// BiMambaBlock_23940147708362
// MI455X (gfx1250) — matrix operand order repaired and checked by reading, not compiled, not run on hardware
//
#include <hip/hip_runtime.h>

// ---------------------------------------------------------------------------
// BiMamba block for MI455X (gfx1250). bf16 WMMA (16x64 tile per wave, K
// fully unrolled) for all large GEMMs, fp32 register-resident selective
// scan, wave32 throughout.
// ---------------------------------------------------------------------------

typedef __attribute__((ext_vector_type(16))) __bf16 v16bf;
typedef __attribute__((ext_vector_type(8)))  float  v8f;

#define BN   4
#define LSEQ 1024
#define CM   192          // D_MODEL
#define DI   384          // D_INNER
#define XZC  768          // 2*D_INNER
#define XDC  44           // DT_RANK + 2*D_STATE
#define DTR  12
#define DST  16
#define BL   (BN*LSEQ)    // 4096 rows

__device__ __forceinline__ float wave_sum(float v) {
    #pragma unroll
    for (int o = 16; o > 0; o >>= 1) v += __shfl_xor(v, o, 32);
    return v;
}
__device__ __forceinline__ float silu_f(float x) {
    return x / (1.0f + __expf(-x));
}

// ---- Kernel 1: LayerNorm over channels, emit bf16 fwd + reversed copies ----
// one wave per (b,t) row; lane handles 6 channels
__global__ void k_ln_prep(const float* __restrict__ x,
                          const float* __restrict__ g, const float* __restrict__ bta,
                          __bf16* __restrict__ xnf, __bf16* __restrict__ xnr) {
    int wave = blockIdx.x * (blockDim.x >> 5) + (threadIdx.x >> 5);
    int lane = threadIdx.x & 31;
    if (wave >= BL) return;
    int b = wave / LSEQ, t = wave % LSEQ;
    float v[6];
    #pragma unroll
    for (int j = 0; j < 6; ++j) {
        int c = lane + 32 * j;
        v[j] = x[((size_t)b * CM + c) * LSEQ + t];
    }
    float s = 0.f;
    #pragma unroll
    for (int j = 0; j < 6; ++j) s += v[j];
    float mean = wave_sum(s) * (1.0f / CM);
    float q = 0.f;
    #pragma unroll
    for (int j = 0; j < 6; ++j) { float d = v[j] - mean; q += d * d; }
    float rstd = rsqrtf(wave_sum(q) * (1.0f / CM) + 1e-5f);
    size_t rf = (size_t)wave * CM;
    size_t rr = ((size_t)b * LSEQ + (LSEQ - 1 - t)) * CM;
    #pragma unroll
    for (int j = 0; j < 6; ++j) {
        int c = lane + 32 * j;
        float nv = (v[j] - mean) * rstd * g[c] + bta[c];
        __bf16 bv = (__bf16)nv;
        xnf[rf + c] = bv;
        xnr[rr + c] = bv;
    }
}

// ---- Kernel 2: f32 -> bf16 convert (weights) ----
__global__ void k_cvt_bf16(const float* __restrict__ in, __bf16* __restrict__ out, int n) {
    int i = blockIdx.x * blockDim.x + threadIdx.x;
    if (i < n) out[i] = (__bf16)in[i];
}

// ---- Kernel 3: A = -exp(A_log) ----
__global__ void k_negexp(const float* __restrict__ in, float* __restrict__ out, int n) {
    int i = blockIdx.x * blockDim.x + threadIdx.x;
    if (i < n) out[i] = -__expf(in[i]);
}

// ---- Kernel 4: bf16 WMMA GEMM  C[M,N] = A[M,K] * W[N,K]^T (+bias) ----
// One wave computes a 16x64 tile (1 M-tile x 4 N-tiles): the A fragment is
// reused by 4 v_wmma_f32_16x16x32_bf16 per K-step. K is a template constant
// so the K-loop fully unrolls and loads pipeline ahead of the WMMAs.
template <int K>
__global__ void k_gemm_bf16(const __bf16* __restrict__ A, const __bf16* __restrict__ W,
                            const float* __restrict__ bias, float* __restrict__ C,
                            int M, int N) {
    int wave = blockIdx.x * (blockDim.x >> 5) + (threadIdx.x >> 5);
    int lane = threadIdx.x & 31;
    int tilesN = N >> 6;                 // 64-wide N tiles
    int totalTiles = (M >> 4) * tilesN;
    if (wave >= totalTiles) return;
    int tm = wave / tilesN, tn = wave % tilesN;
    int half = lane >> 4;                // lane half selects K sub-range
    int r    = lane & 15;
    const __bf16* arow = A + (size_t)(tm * 16 + r) * K;
    const __bf16* wrow[4];
    #pragma unroll
    for (int j = 0; j < 4; ++j)
        wrow[j] = W + (size_t)(tn * 64 + j * 16 + r) * K;
    v8f acc[4] = {};
    #pragma unroll
    for (int k0 = 0; k0 < K; k0 += 32) {
        __builtin_prefetch(arow + k0 + 64, 0, 0);       // global_prefetch_b8
        // A fragment, contiguous k order chosen here (B matches): lanes 0-15 K=0..15, lanes 16-31 K=16..31
        v16bf a;
        int ab = k0 + half * 16;
        #pragma unroll
        for (int i = 0; i < 8; ++i) a[i]     = arow[ab + i];
        #pragma unroll
        for (int i = 0; i < 8; ++i) a[8 + i] = arow[ab + 8 + i];
        // B fragment, same contiguous k order as A: lanes 0-15 hold K=0..15, lanes 16-31 K=16..31
        int bb = k0 + half * 16;
        v16bf bfr[4];
        #pragma unroll
        for (int j = 0; j < 4; ++j)
            #pragma unroll
            for (int i = 0; i < 16; ++i) bfr[j][i] = wrow[j][bb + i];
        #pragma unroll
        for (int j = 0; j < 4; ++j)
            acc[j] = __builtin_amdgcn_wmma_f32_16x16x32_bf16(false, a, false, bfr[j],
                                                             (short)0, acc[j], false, false);
    }
    // C/D f32 16x16 layout: VGPR v -> row (half*8 + v), col = lane&15
    int row0 = tm * 16 + half * 8;
    #pragma unroll
    for (int j = 0; j < 4; ++j) {
        int col = tn * 64 + j * 16 + r;
        float bv = bias ? bias[col] : 0.f;
        #pragma unroll
        for (int v = 0; v < 8; ++v) C[(size_t)(row0 + v) * N + col] = acc[j][v] + bv;
    }
}

// ---- Kernel 5: depthwise causal conv(4) + bias + SiLU ----
__global__ void k_conv_silu(const float* __restrict__ xz, const float* __restrict__ cw,
                            const float* __restrict__ cb, float* __restrict__ u) {
    int idx = blockIdx.x * blockDim.x + threadIdx.x;
    if (idx >= BL * DI) return;
    int c = idx % DI;
    int row = idx / DI;            // b*LSEQ + t
    int b = row / LSEQ, t = row % LSEQ;
    float acc = cb[c];
    #pragma unroll
    for (int k = 0; k < 4; ++k) {
        int ts = t - 3 + k;
        if (ts >= 0)
            acc += cw[c * 4 + k] * xz[((size_t)b * LSEQ + ts) * XZC + c];
    }
    u[(size_t)row * DI + c] = silu_f(acc);
}

// ---- Kernel 6: x_dbl = u @ xp_w^T  (N=44, K=384; plain VALU dot) ----
__global__ void k_xdbl(const float* __restrict__ u, const float* __restrict__ xp_w,
                       float* __restrict__ xdbl) {
    int idx = blockIdx.x * blockDim.x + threadIdx.x;
    if (idx >= BL * XDC) return;
    int col = idx % XDC, row = idx / XDC;
    const float* ur = u + (size_t)row * DI;
    const float* wr = xp_w + (size_t)col * DI;
    float acc = 0.f;
    #pragma unroll 4
    for (int k = 0; k < DI; ++k) acc += ur[k] * wr[k];
    xdbl[(size_t)row * XDC + col] = acc;
}

// ---- Kernel 7: selective scan. one lane per (b, d); h[16] in VGPRs ----
__global__ void k_scan(const float* __restrict__ xdbl, const float* __restrict__ u,
                       const float* __restrict__ xz, const float* __restrict__ dt_w,
                       const float* __restrict__ dt_b, const float* __restrict__ Aneg,
                       const float* __restrict__ Dp, __bf16* __restrict__ y) {
    int gid = blockIdx.x * blockDim.x + threadIdx.x;
    if (gid >= BN * DI) return;
    int b = gid / DI, d = gid % DI;
    float dtw[DTR], Arow[DST], h[DST];
    #pragma unroll
    for (int k = 0; k < DTR; ++k) dtw[k] = dt_w[d * DTR + k];
    #pragma unroll
    for (int s = 0; s < DST; ++s) { Arow[s] = Aneg[d * DST + s]; h[s] = 0.f; }
    float dtb = dt_b[d], dcoef = Dp[d];
    for (int t = 0; t < LSEQ; ++t) {
        size_t row = (size_t)b * LSEQ + t;
        const float* xr = xdbl + row * XDC;
        float dl = dtb;
        #pragma unroll
        for (int k = 0; k < DTR; ++k) dl += dtw[k] * xr[k];
        dl = (dl > 20.f) ? dl : log1pf(__expf(dl));      // softplus
        float uu = u[row * DI + d];
        float du = dl * uu;
        float yacc = 0.f;
        #pragma unroll
        for (int s = 0; s < DST; ++s) {
            float dA = __expf(dl * Arow[s]);
            h[s] = dA * h[s] + du * xr[DTR + s];         // Bs
            yacc += h[s] * xr[DTR + DST + s];            // Cs
        }
        float z = xz[row * XZC + DI + d];
        y[row * DI + d] = (__bf16)((yacc + uu * dcoef) * silu_f(z));
    }
}

// ---- Kernel 8: combine fwd + flipped bwd into bf16 (4096 x 384) ----
__global__ void k_combine(const float* __restrict__ of, const float* __restrict__ ob,
                          __bf16* __restrict__ comb) {
    int idx = blockIdx.x * blockDim.x + threadIdx.x;
    if (idx >= BL * CM) return;
    int c = idx % CM, row = idx / CM;
    int b = row / LSEQ, t = row % LSEQ;
    comb[(size_t)row * (2 * CM) + c]      = (__bf16)of[(size_t)row * CM + c];
    comb[(size_t)row * (2 * CM) + CM + c] =
        (__bf16)ob[((size_t)b * LSEQ + (LSEQ - 1 - t)) * CM + c];
}

// ---- Kernel 9: final LN + residual, scatter back to NCHW ----
__global__ void k_final(const float* __restrict__ merged, const float* __restrict__ g,
                        const float* __restrict__ bta, const float* __restrict__ x,
                        float* __restrict__ out) {
    int wave = blockIdx.x * (blockDim.x >> 5) + (threadIdx.x >> 5);
    int lane = threadIdx.x & 31;
    if (wave >= BL) return;
    int b = wave / LSEQ, t = wave % LSEQ;
    float v[6];
    #pragma unroll
    for (int j = 0; j < 6; ++j) v[j] = merged[(size_t)wave * CM + lane + 32 * j];
    float s = 0.f;
    #pragma unroll
    for (int j = 0; j < 6; ++j) s += v[j];
    float mean = wave_sum(s) * (1.0f / CM);
    float q = 0.f;
    #pragma unroll
    for (int j = 0; j < 6; ++j) { float d = v[j] - mean; q += d * d; }
    float rstd = rsqrtf(wave_sum(q) * (1.0f / CM) + 1e-5f);
    #pragma unroll
    for (int j = 0; j < 6; ++j) {
        int c = lane + 32 * j;
        size_t xi = ((size_t)b * CM + c) * LSEQ + t;
        out[xi] = x[xi] + (v[j] - mean) * rstd * g[c] + bta[c];
    }
}

// ---------------------------------------------------------------------------
extern "C" void kernel_launch(void* const* d_in, const int* in_sizes, int n_in,
                              void* d_out, int out_size, void* d_ws, size_t ws_size,
                              hipStream_t stream) {
    const float* x      = (const float*)d_in[0];
    const float* n1_g   = (const float*)d_in[1];
    const float* n1_b   = (const float*)d_in[2];
    const float* in_w[2]   = { (const float*)d_in[3],  (const float*)d_in[12] };
    const float* conv_w[2] = { (const float*)d_in[4],  (const float*)d_in[13] };
    const float* conv_b[2] = { (const float*)d_in[5],  (const float*)d_in[14] };
    const float* xp_w[2]   = { (const float*)d_in[6],  (const float*)d_in[15] };
    const float* dt_w[2]   = { (const float*)d_in[7],  (const float*)d_in[16] };
    const float* dt_b[2]   = { (const float*)d_in[8],  (const float*)d_in[17] };
    const float* A_log[2]  = { (const float*)d_in[9],  (const float*)d_in[18] };
    const float* Dp[2]     = { (const float*)d_in[10], (const float*)d_in[19] };
    const float* out_w[2]  = { (const float*)d_in[11], (const float*)d_in[20] };
    const float* lp_w   = (const float*)d_in[21];
    const float* lp_b   = (const float*)d_in[22];
    const float* n2_g   = (const float*)d_in[23];
    const float* n2_b   = (const float*)d_in[24];

    // bump allocator on workspace (256B aligned)
    char* base = (char*)d_ws; size_t off = 0;
    auto alloc = [&](size_t bytes) -> void* {
        void* r = base + off;
        off = (off + bytes + 255) & ~(size_t)255;
        return r;
    };
    __bf16* xn[2]    = { (__bf16*)alloc((size_t)BL*CM*2), (__bf16*)alloc((size_t)BL*CM*2) };
    __bf16* w_in[2]  = { (__bf16*)alloc((size_t)XZC*CM*2), (__bf16*)alloc((size_t)XZC*CM*2) };
    __bf16* w_out[2] = { (__bf16*)alloc((size_t)CM*DI*2), (__bf16*)alloc((size_t)CM*DI*2) };
    __bf16* w_lp     =   (__bf16*)alloc((size_t)CM*(2*CM)*2);
    float*  Aneg[2]  = { (float*)alloc((size_t)DI*DST*4), (float*)alloc((size_t)DI*DST*4) };
    float*  xz[2]    = { (float*)alloc((size_t)BL*XZC*4), (float*)alloc((size_t)BL*XZC*4) };
    float*  u[2]     = { (float*)alloc((size_t)BL*DI*4),  (float*)alloc((size_t)BL*DI*4) };
    float*  xdbl[2]  = { (float*)alloc((size_t)BL*XDC*4), (float*)alloc((size_t)BL*XDC*4) };
    __bf16* ybf[2]   = { (__bf16*)alloc((size_t)BL*DI*2), (__bf16*)alloc((size_t)BL*DI*2) };
    float*  oproj[2] = { (float*)alloc((size_t)BL*CM*4),  (float*)alloc((size_t)BL*CM*4) };
    __bf16* comb     =   (__bf16*)alloc((size_t)BL*(2*CM)*2);
    float*  merged   =   (float*)alloc((size_t)BL*CM*4);

    auto gemm = [&](const __bf16* A, const __bf16* W, const float* bias, float* C,
                    int M, int N, int K) {
        int tiles = (M / 16) * (N / 64);
        int blocks = (tiles + 7) / 8;
        if (K == CM)
            k_gemm_bf16<CM><<<dim3(blocks), dim3(256), 0, stream>>>(A, W, bias, C, M, N);
        else
            k_gemm_bf16<DI><<<dim3(blocks), dim3(256), 0, stream>>>(A, W, bias, C, M, N);
    };

    // 1) LayerNorm + bf16 fwd/rev copies  (one wave per row)
    k_ln_prep<<<dim3(BL / 4), dim3(128), 0, stream>>>(x, n1_g, n1_b, xn[0], xn[1]);

    // 2) weight conversions + A = -exp(A_log)
    for (int d = 0; d < 2; ++d) {
        k_cvt_bf16<<<dim3((XZC*CM + 255)/256), dim3(256), 0, stream>>>(in_w[d],  w_in[d],  XZC*CM);
        k_cvt_bf16<<<dim3((CM*DI + 255)/256),  dim3(256), 0, stream>>>(out_w[d], w_out[d], CM*DI);
        k_negexp  <<<dim3((DI*DST + 255)/256), dim3(256), 0, stream>>>(A_log[d], Aneg[d],  DI*DST);
    }
    k_cvt_bf16<<<dim3((CM*2*CM + 255)/256), dim3(256), 0, stream>>>(lp_w, w_lp, CM*2*CM);

    // 3) per-direction mamba
    for (int d = 0; d < 2; ++d) {
        gemm(xn[d], w_in[d], nullptr, xz[d], BL, XZC, CM);                  // in_proj (WMMA)
        k_conv_silu<<<dim3((BL*DI + 255)/256), dim3(256), 0, stream>>>(xz[d], conv_w[d], conv_b[d], u[d]);
        k_xdbl<<<dim3((BL*XDC + 255)/256), dim3(256), 0, stream>>>(u[d], xp_w[d], xdbl[d]);
        k_scan<<<dim3((BN*DI + 255)/256), dim3(256), 0, stream>>>(xdbl[d], u[d], xz[d],
                 dt_w[d], dt_b[d], Aneg[d], Dp[d], ybf[d]);
        gemm(ybf[d], w_out[d], nullptr, oproj[d], BL, CM, DI);              // out_proj (WMMA)
    }

    // 4) combine (un-flip bwd), merge GEMM, final LN + residual
    k_combine<<<dim3((BL*CM + 255)/256), dim3(256), 0, stream>>>(oproj[0], oproj[1], comb);
    gemm(comb, w_lp, lp_b, merged, BL, CM, 2*CM);                           // merge (WMMA)
    k_final<<<dim3(BL / 4), dim3(128), 0, stream>>>(merged, n2_g, n2_b, x, (float*)d_out);
}
